// SPVCNN34_cls_28544352649669
// MI455X (gfx1250) — matrix operand order repaired and checked by reading, not compiled, not run on hardware
//
#include <hip/hip_runtime.h>

typedef _Float16 f16;
typedef __attribute__((ext_vector_type(16))) _Float16 v16h;
typedef __attribute__((ext_vector_type(8)))  _Float16 v8h;
typedef __attribute__((ext_vector_type(8)))  float    v8f;

#define EPSV 1e-5f

// -------------------------------------------------------------------------
// utility: zero a float buffer
// -------------------------------------------------------------------------
__global__ void k_zero(float* __restrict__ p, int n) {
  int i = blockIdx.x * blockDim.x + threadIdx.x;
  if (i < n) p[i] = 0.f;
}

// -------------------------------------------------------------------------
// voxelize: scatter-add point features + counts (mean computed in k_vox_div)
// -------------------------------------------------------------------------
__global__ void k_vox_scatter(const float* __restrict__ pf,
                              const float* __restrict__ coords,
                              float* __restrict__ sums,
                              float* __restrict__ counts,
                              int C, int N) {
  int n = blockIdx.x * blockDim.x + threadIdx.x;
  if (n >= N) return;
  float px = coords[n*4+0], py = coords[n*4+1], pz = coords[n*4+2];
  int b = (int)coords[n*4+3];
  int ix = (int)floorf(px); ix = ix < 0 ? 0 : (ix > 31 ? 31 : ix);
  int iy = (int)floorf(py); iy = iy < 0 ? 0 : (iy > 31 ? 31 : iy);
  int iz = (int)floorf(pz); iz = iz < 0 ? 0 : (iz > 31 ? 31 : iz);
  int seg = (((b*32+ix)*32+iy)*32+iz);
  const float* f = pf + (long)n * C;
  float* s = sums + (long)seg * C;
  for (int c = 0; c < C; ++c) atomicAdd(&s[c], f[c]);
  atomicAdd(&counts[seg], 1.f);
}

__global__ void k_vox_div(float* __restrict__ grid,
                          const float* __restrict__ sums,
                          const float* __restrict__ counts,
                          int n, int lc) {
  int i = blockIdx.x * blockDim.x + threadIdx.x;
  if (i >= n) return;
  float c = counts[i >> lc];
  grid[i] = sums[i] / fmaxf(c, 1.f);
}

// -------------------------------------------------------------------------
// fused devoxelize (trilinear gather, C=32) + voxelize scatter back to grid.
// Avoids materializing the 32 MB per-point feature tensor entirely.
// -------------------------------------------------------------------------
__global__ void k_devox_vox(const float* __restrict__ grid,
                            const float* __restrict__ coords,
                            float* __restrict__ sums,
                            float* __restrict__ counts, int N) {
  int n = blockIdx.x * blockDim.x + threadIdx.x;
  if (n >= N) return;
  float px = coords[n*4+0], py = coords[n*4+1], pz = coords[n*4+2];
  int b = (int)coords[n*4+3];
  float bx = floorf(px), by = floorf(py), bz = floorf(pz);
  float fx = px-bx, fy = py-by, fz = pz-bz;
  int ix = (int)bx, iy = (int)by, iz = (int)bz;

  int idx8[8]; float w8[8];
#pragma unroll
  for (int cn = 0; cn < 8; ++cn) {
    int dx = (cn>>2)&1, dy = (cn>>1)&1, dz = cn&1;
    int cx = ix+dx; cx = cx < 0 ? 0 : (cx > 31 ? 31 : cx);
    int cy = iy+dy; cy = cy < 0 ? 0 : (cy > 31 ? 31 : cy);
    int cz = iz+dz; cz = cz < 0 ? 0 : (cz > 31 ? 31 : cz);
    idx8[cn] = ((((b*32+cx)*32+cy)*32+cz)) * 32;
    w8[cn] = (dx ? fx : 1.f-fx) * (dy ? fy : 1.f-fy) * (dz ? fz : 1.f-fz);
  }
  int sx = ix < 0 ? 0 : (ix > 31 ? 31 : ix);
  int sy = iy < 0 ? 0 : (iy > 31 ? 31 : iy);
  int sz = iz < 0 ? 0 : (iz > 31 ? 31 : iz);
  int seg = (((b*32+sx)*32+sy)*32+sz);
  float* s = sums + (long)seg * 32;

#pragma unroll
  for (int c0 = 0; c0 < 32; c0 += 4) {
    float a0 = 0.f, a1 = 0.f, a2 = 0.f, a3 = 0.f;
#pragma unroll
    for (int cn = 0; cn < 8; ++cn) {
      const float4 v = *(const float4*)(grid + idx8[cn] + c0);
      float w = w8[cn];
      a0 += w*v.x; a1 += w*v.y; a2 += w*v.z; a3 += w*v.w;
    }
    atomicAdd(&s[c0+0], a0); atomicAdd(&s[c0+1], a1);
    atomicAdd(&s[c0+2], a2); atomicAdd(&s[c0+3], a3);
  }
  atomicAdd(&counts[seg], 1.f);
}

// -------------------------------------------------------------------------
// Implicit-GEMM conv3d via V_WMMA_F32_16X16X32_F16 (wave32, 1 wave/block).
// Register-blocked 2x2: each wave computes M = 32 flattened output voxels
// x N = 32 output channels as four 16x16 WMMA tiles, so each staged A/B
// fragment is used twice and each K-chunk issues 4 v_wmma ops.
// K = ntaps*Cin in chunks of 32 (Cin is a power of two).
// mode 0: 3x3x3 SAME,  mode 1: 2x2x2 stride-2 VALID,  mode 2: 1x1x1.
// Also accumulates per-channel sum / sum-of-squares for batch-norm.
// -------------------------------------------------------------------------
__global__ __launch_bounds__(32)
void k_conv_wmma(const float* __restrict__ in, const float* __restrict__ wgt,
                 float* __restrict__ out, float* __restrict__ st,
                 int lgi, int lgo, int Cin, int lc, int Cout, int mode,
                 int lnT, int lsT) {
  const int Gi = 1 << lgi;
  const int Go = 1 << lgo;
  const int ntaps = (mode == 0) ? 27 : ((mode == 1) ? 8 : 1);
  const int Kdim = ntaps << lc;
  const int lane = threadIdx.x;
  int t = blockIdx.x;                       // power-of-two decode: no int division
  const int nt  = t & ((1 << lnT) - 1); t >>= lnT;
  const int stt = t & ((1 << lsT) - 1);
  const int b   = t >> lsT;
  const int co0 = nt << 5;
  const int s0  = stt << 5;

  __shared__ __align__(16) f16 As[1024];   // A tile [M=32][K=32]
  __shared__ __align__(16) f16 Bs[1024];   // B tile transposed [N=32][K=32]

  v8f acc00 = {0.f,0.f,0.f,0.f,0.f,0.f,0.f,0.f};
  v8f acc01 = {0.f,0.f,0.f,0.f,0.f,0.f,0.f,0.f};
  v8f acc10 = {0.f,0.f,0.f,0.f,0.f,0.f,0.f,0.f};
  v8f acc11 = {0.f,0.f,0.f,0.f,0.f,0.f,0.f,0.f};
  const int hf = lane >> 4;
  const int l  = lane & 15;

  for (int kk = 0; kk < Kdim; kk += 32) {
    const int  kg = kk + lane;
    const bool kv = kg < Kdim;
    const int  tap = kg >> lc;
    const int  ci  = kg & (Cin - 1);
    int dx = 0, dy = 0, dz = 0;
    if (mode == 0)      { dx = tap/9 - 1; dy = (tap/3)%3 - 1; dz = tap%3 - 1; }
    else if (mode == 1) { dx = (tap>>2)&1; dy = (tap>>1)&1; dz = tap&1; }

    // prefetch next K-chunk's weight row into cache (global_prefetch_b8)
    if (kk + 32 < Kdim) __builtin_prefetch(&wgt[(kg + 32)*Cout + co0], 0, 1);

    // B tile: weights already live as [K][Cout] (DHWIO) -> coalesced reads
#pragma unroll
    for (int n = 0; n < 32; ++n) {
      float wv = kv ? wgt[kg*Cout + co0 + n] : 0.f;
      Bs[n*32 + lane] = (f16)wv;
    }
    // A tile: one coalesced channel-vector read per output voxel row
#pragma unroll
    for (int m = 0; m < 32; ++m) {
      const int s  = s0 + m;
      const int vz = s & (Go-1);
      const int vy = (s >> lgo) & (Go-1);
      const int vx = s >> (2*lgo);
      float av = 0.f;
      if (kv) {
        int xi, yi, zi; bool ok = true;
        if (mode == 0) {
          xi = vx+dx; yi = vy+dy; zi = vz+dz;
          ok = (unsigned)xi < (unsigned)Gi && (unsigned)yi < (unsigned)Gi &&
               (unsigned)zi < (unsigned)Gi;
        } else if (mode == 1) {
          xi = (vx<<1)+dx; yi = (vy<<1)+dy; zi = (vz<<1)+dz;
        } else { xi = vx; yi = vy; zi = vz; }
        if (ok) {
          int vox = (b << lgi) + xi;
          vox = (vox << lgi) + yi;
          vox = (vox << lgi) + zi;
          av = in[(long)vox * Cin + ci];
        }
      }
      As[m*32 + lane] = (f16)av;
    }
    __syncthreads();

    // A fragments: lanes 0-15 hold M=l, K 0-15; lanes 16-31 K 16-31
    const f16* pa0 = &As[l*32 + hf*16];
    const f16* pa1 = &As[(16 + l)*32 + hf*16];
    v16h a0, a1;
    {
      const v8h lo0 = *(const v8h*)pa0, hi0 = *(const v8h*)(pa0 + 8);
      const v8h lo1 = *(const v8h*)pa1, hi1 = *(const v8h*)(pa1 + 8);
#pragma unroll
      for (int i = 0; i < 8; ++i) {
        a0[i] = lo0[i]; a0[8+i] = hi0[i];
        a1[i] = lo1[i]; a1[8+i] = hi1[i];
      }
    }
    // B fragments: column N, 16 contiguous K values (hf selects K half)
    const v16h b0 = *(const v16h*)(&Bs[l*32 + hf*16]);
    const v16h b1 = *(const v16h*)(&Bs[(16 + l)*32 + hf*16]);

    acc00 = __builtin_amdgcn_wmma_f32_16x16x32_f16(false, a0, false, b0,
                                                   (short)0, acc00, false, false);
    acc01 = __builtin_amdgcn_wmma_f32_16x16x32_f16(false, a0, false, b1,
                                                   (short)0, acc01, false, false);
    acc10 = __builtin_amdgcn_wmma_f32_16x16x32_f16(false, a1, false, b0,
                                                   (short)0, acc10, false, false);
    acc11 = __builtin_amdgcn_wmma_f32_16x16x32_f16(false, a1, false, b1,
                                                   (short)0, acc11, false, false);
    __syncthreads();
  }

  // D layout: VGPR r -> M = r + 8*hf, N = l. Store + BN statistics.
  const int spat = Go*Go*Go;
  float s00 = 0.f, q00 = 0.f, s01 = 0.f, q01 = 0.f;
#pragma unroll
  for (int r = 0; r < 8; ++r) {
    const int m = r + (hf << 3);
    const long base0 = (long)(b*spat + s0 + m)      * Cout;
    const long base1 = (long)(b*spat + s0 + 16 + m) * Cout;
    float v;
    v = acc00[r]; out[base0 + co0 + l]      = v; s00 += v; q00 += v*v;
    v = acc01[r]; out[base0 + co0 + 16 + l] = v; s01 += v; q01 += v*v;
    v = acc10[r]; out[base1 + co0 + l]      = v; s00 += v; q00 += v*v;
    v = acc11[r]; out[base1 + co0 + 16 + l] = v; s01 += v; q01 += v*v;
  }
  atomicAdd(&st[co0 + l],            s00);
  atomicAdd(&st[256 + co0 + l],      q00);
  atomicAdd(&st[co0 + 16 + l],       s01);
  atomicAdd(&st[256 + co0 + 16 + l], q01);
}

// -------------------------------------------------------------------------
// batch-norm finalize: per-channel scale/bias from the conv's sum/sumsq
// -------------------------------------------------------------------------
__global__ void k_finalize(float* __restrict__ st, const float* __restrict__ g,
                           const float* __restrict__ b, int C, float invcnt) {
  int c = threadIdx.x;
  if (c >= C) return;
  float m  = st[c] * invcnt;
  float v  = st[256 + c] * invcnt - m*m;
  float sc = g[c] * rsqrtf(v + EPSV);
  st[512 + c] = sc;
  st[768 + c] = b[c] - m*sc;
}

// -------------------------------------------------------------------------
// elementwise epilogue: y = act( norm(a) [+ norm(a2)] [+ res] )
// -------------------------------------------------------------------------
__global__ void k_apply(float* __restrict__ y, const float* __restrict__ a,
                        const float* __restrict__ sA, const float* __restrict__ bA,
                        const float* __restrict__ a2, const float* __restrict__ sB,
                        const float* __restrict__ bB, const float* __restrict__ res,
                        int n, int cmask, int relu) {
  int i = blockIdx.x * blockDim.x + threadIdx.x;
  if (i >= n) return;
  int c = i & cmask;
  float v = fmaf(a[i], sA[c], bA[c]);
  if (a2)  v += fmaf(a2[i], sB[c], bB[c]);
  if (res) v += res[i];
  if (relu) v = fmaxf(v, 0.f);
  y[i] = v;
}

// -------------------------------------------------------------------------
// head: spatial mean -> 256x40 classifier -> batch-norm over batch (B=2)
// -------------------------------------------------------------------------
__global__ __launch_bounds__(256)
void k_head(const float* __restrict__ x4, const float* __restrict__ w,
            const float* __restrict__ cb, const float* __restrict__ g,
            const float* __restrict__ beta, float* __restrict__ out) {
  __shared__ float x5[512];
  __shared__ float h[80];
  const int tid = threadIdx.x;
  for (int idx = tid; idx < 512; idx += 256) {
    const int b = idx >> 8, c = idx & 255;
    float s = 0.f;
    for (int v = 0; v < 64; ++v) s += x4[(b*64 + v)*256 + c];
    x5[idx] = s * (1.f/64.f);
  }
  __syncthreads();
  if (tid < 80) {
    const int b = tid/40, n = tid%40;
    float acc = cb[n];
    for (int c = 0; c < 256; ++c) acc = fmaf(x5[b*256+c], w[c*40+n], acc);
    h[tid] = acc;
  }
  __syncthreads();
  if (tid < 40) {
    const float h0 = h[tid], h1 = h[40+tid];
    const float m  = 0.5f*(h0+h1);
    const float d0 = h0-m, d1 = h1-m;
    const float v  = 0.5f*(d0*d0 + d1*d1);
    const float sc = rsqrtf(v + EPSV) * g[tid];
    out[tid]    = d0*sc + beta[tid];
    out[40+tid] = d1*sc + beta[tid];
  }
}

// =========================================================================
extern "C" void kernel_launch(void* const* d_in, const int* in_sizes, int n_in,
                              void* d_out, int out_size, void* d_ws, size_t ws_size,
                              hipStream_t stream) {
  (void)in_sizes; (void)n_in; (void)out_size; (void)ws_size;
  const int N = 262144, Bn = 2;

  int pi = 0;
  auto nx = [&]() -> const float* { return (const float*)d_in[pi++]; };
  const float* feats  = nx();
  const float* coords = nx();
  const float* stem_w = nx();
  const float* stem_g = nx();
  const float* stem_b = nx();

  struct RBp { const float *w1,*g1,*b1,*w2,*g2,*b2,*wd,*gd,*bd; };
  auto rbp = [&](bool wd) {
    RBp p{};
    p.w1 = nx(); p.g1 = nx(); p.b1 = nx();
    p.w2 = nx(); p.g2 = nx(); p.b2 = nx();
    if (wd) { p.wd = nx(); p.gd = nx(); p.bd = nx(); }
    return p;
  };
  RBp s1[2]; s1[0] = rbp(false); s1[1] = rbp(false);
  const float *d1w = nx(), *d1g = nx(), *d1b = nx();
  RBp s2[3]; s2[0] = rbp(true); s2[1] = rbp(false); s2[2] = rbp(false);
  const float *d2w = nx(), *d2g = nx(), *d2b = nx();
  RBp s3[4]; s3[0] = rbp(true); for (int i = 1; i < 4; ++i) s3[i] = rbp(false);
  const float *d3w = nx(), *d3g = nx(), *d3b = nx();
  RBp s4[6]; s4[0] = rbp(true); for (int i = 1; i < 6; ++i) s4[i] = rbp(false);
  const float *clsw = nx(), *clsb = nx(), *clsg = nx(), *clsbeta = nx();

  // workspace layout (floats): 2 BN stat slots, counts, 4 ping-pong buffers
  float* ws     = (float*)d_ws;
  float* st0    = ws;                       // 1024 (sum,sq,scale,bias @ +0/256/512/768)
  float* st1    = ws + 1024;                // 1024
  float* counts = ws + 2048;                // 65536
  float* bufA   = ws + 2048 + 65536;        // 4 x 2M floats (8 MB each)
  float* bufB   = bufA + (1 << 21);
  float* bufC   = bufB + (1 << 21);
  float* bufD   = bufC + (1 << 21);

  auto zero = [&](float* p, int n) {
    k_zero<<<dim3((n+255)/256), dim3(256), 0, stream>>>(p, n);
  };
  auto conv = [&](const float* in, const float* w, float* o, float* st,
                  int lgi, int lgo, int Cin, int Cout, int mode) {
    int lc = __builtin_ctz(Cin);
    int Go = 1 << lgo;
    int spatial = Go*Go*Go;
    int lsT = __builtin_ctz(spatial >> 5);
    int lnT = __builtin_ctz(Cout >> 5);
    int blocks = Bn * (spatial >> 5) * (Cout >> 5);
    k_conv_wmma<<<dim3(blocks), dim3(32), 0, stream>>>(in, w, o, st,
                                                       lgi, lgo, Cin, lc, Cout,
                                                       mode, lnT, lsT);
  };
  auto finalize = [&](float* st, const float* g, const float* b, int C, int cnt) {
    k_finalize<<<dim3(1), dim3(256), 0, stream>>>(st, g, b, C, 1.f/(float)cnt);
  };
  auto apply = [&](float* y, const float* a, float* stA,
                   const float* a2, float* stB, const float* res,
                   int n, int C, int relu) {
    k_apply<<<dim3((n+255)/256), dim3(256), 0, stream>>>(
        y, a, stA + 512, stA + 768,
        a2, stB ? stB + 512 : nullptr, stB ? stB + 768 : nullptr,
        res, n, C - 1, relu);
  };

  // ---- voxelize feats (C=4, 32^3 grid) ----
  zero(bufA, 65536*4); zero(counts, 65536);
  k_vox_scatter<<<dim3((N+255)/256), dim3(256), 0, stream>>>(feats, coords, bufA, counts, 4, N);
  k_vox_div<<<dim3((65536*4+255)/256), dim3(256), 0, stream>>>(bufB, bufA, counts, 65536*4, 2);

  // ---- stem conv (4->32) + BN (no relu) ----
  zero(st0, 512);
  conv(bufB, stem_w, bufC, st0, 5, 5, 4, 32, 0);
  finalize(st0, stem_g, stem_b, 32, 65536);
  apply(bufD, bufC, st0, nullptr, nullptr, nullptr, 65536*32, 32, 0);

  // ---- fused devoxelize + voxelize back (C=32) ----
  zero(bufA, 65536*32); zero(counts, 65536);
  k_devox_vox<<<dim3((N+255)/256), dim3(256), 0, stream>>>(bufD, coords, bufA, counts, N);
  k_vox_div<<<dim3((65536*32+255)/256), dim3(256), 0, stream>>>(bufB, bufA, counts, 65536*32, 5);

  float* x  = bufB;
  float* f0 = bufA; float* f1 = bufC; float* f2 = bufD;

  auto resblock = [&](const RBp& p, int Cin, int Cout, int lg) {
    int nvox = Bn << (3*lg);
    int n = nvox * Cout;
    zero(st0, 512);
    conv(x, p.w1, f0, st0, lg, lg, Cin, Cout, 0);
    finalize(st0, p.g1, p.b1, Cout, nvox);
    apply(f0, f0, st0, nullptr, nullptr, nullptr, n, Cout, 1);   // h = relu(bn(conv1))
    zero(st0, 512);
    conv(f0, p.w2, f1, st0, lg, lg, Cout, Cout, 0);
    finalize(st0, p.g2, p.b2, Cout, nvox);
    if (p.wd) {
      zero(st1, 512);
      conv(x, p.wd, f2, st1, lg, lg, Cin, Cout, 2);              // 1x1x1 shortcut
      finalize(st1, p.gd, p.bd, Cout, nvox);
      apply(f0, f1, st0, f2, st1, nullptr, n, Cout, 1);          // relu(bn2+bnd)
    } else {
      apply(f0, f1, st0, nullptr, nullptr, x, n, Cout, 1);       // relu(bn2+x)
    }
    float* oldx = x; x = f0; f0 = oldx;
  };

  auto dsop = [&](const float* w, const float* g, const float* b, int C, int lgo) {
    int nvox = Bn << (3*lgo);
    zero(st0, 512);
    conv(x, w, f0, st0, lgo + 1, lgo, C, C, 1);                  // 2x2x2 stride-2
    finalize(st0, g, b, C, nvox);
    apply(f0, f0, st0, nullptr, nullptr, nullptr, nvox*C, C, 1);
    float* oldx = x; x = f0; f0 = oldx;
  };

  resblock(s1[0], 32, 32, 5);
  resblock(s1[1], 32, 32, 5);
  dsop(d1w, d1g, d1b, 32, 4);

  resblock(s2[0], 32, 64, 4);
  resblock(s2[1], 64, 64, 4);
  resblock(s2[2], 64, 64, 4);
  dsop(d2w, d2g, d2b, 64, 3);

  resblock(s3[0], 64, 128, 3);
  resblock(s3[1], 128, 128, 3);
  resblock(s3[2], 128, 128, 3);
  resblock(s3[3], 128, 128, 3);
  dsop(d3w, d3g, d3b, 128, 2);

  resblock(s4[0], 128, 256, 2);
  for (int i = 1; i < 6; ++i) resblock(s4[i], 256, 256, 2);

  k_head<<<dim3(1), dim3(256), 0, stream>>>(x, clsw, clsb, clsg, clsbeta, (float*)d_out);
}
